// Transformer_70935679861213
// MI455X (gfx1250) — compile-verified, matrix operand order checked by reading, not run on hardware
//
#include <hip/hip_runtime.h>

typedef unsigned int   u32;
typedef unsigned short u16;
typedef unsigned long long u64;
typedef __attribute__((ext_vector_type(16))) __bf16 v16bf;
typedef __attribute__((ext_vector_type(8)))  float  v8f;

#define BATCH 128
#define NNODE 1022
#define NTIP  512
#define HDIM  256

// Async global->LDS DMA (CDNA5).  %0 = VGPR with LDS byte offset, %1 = VGPR
// pair with 64-bit global address.  Tracked by ASYNCcnt.
#define ASYNC_B128(ldsoff, gaddr64)                                            \
  asm volatile("global_load_async_to_lds_b128 %0, %1, off"                     \
               :: "v"(ldsoff), "v"(gaddr64) : "memory")

static __device__ __forceinline__ u16 f2bf(float f) {
  union { __bf16 h; u16 u; } c;
  c.h = (__bf16)f;
  return c.u;
}
static __device__ __forceinline__ float bf2f(u16 h) {
  return __uint_as_float(((u32)h) << 16);
}

// -------------------------------------------------------------------------
// Weight convert + transpose: dst (N,K) bf16 row-major = src[k*ld + coloff + n]
// -------------------------------------------------------------------------
__global__ void convw_kernel(const float* __restrict__ src, u16* __restrict__ dst,
                             int K, int N, int ld, int coloff) {
  int idx = blockIdx.x * 256 + threadIdx.x;
  if (idx >= K * N) return;
  int n = idx / K;
  int k = idx - n * K;
  dst[(size_t)n * K + k] = f2bf(src[(size_t)k * ld + coloff + n]);
}

// -------------------------------------------------------------------------
// WMMA GEMM: C[M,N] = epi(A[M,K] * Bt[N,K]^T + bias)
//  Block: 128 threads = 4 waves; block tile 128x64; wave tile 32x64.
//  Per K-step (32): each wave -> 2 A frags x 4 B frags = 8 WMMAs.
//  A_F32: A is f32 (VGPR convert path); else A staged via async DMA.
//  B always staged via async DMA.  Double-buffered LDS (2 x 12KB).
//  EPI 0: f32 raw out.  EPI 1: ELU then bf16 out.
// -------------------------------------------------------------------------
__device__ __forceinline__ v16bf load_frag(const u32* __restrict__ smem, int row, int hi) {
  union { v16bf v; u32 u[8]; } f;
  const u32* p = smem + row * 16 + hi * 4;
#pragma unroll
  for (int i = 0; i < 4; i++) f.u[i] = p[i];
  p += 8;
#pragma unroll
  for (int i = 0; i < 4; i++) f.u[4 + i] = p[i];
  return f.v;
}

template <bool A_F32, int EPI>
__global__ __launch_bounds__(128) void gemm_wmma(const void* __restrict__ Aptr,
                                                 const u16* __restrict__ Bt,
                                                 const float* __restrict__ bias,
                                                 void* __restrict__ Cptr,
                                                 int K, int Ncols) {
  __shared__ u32 As[2][128 * 16];  // 128 rows x 32 bf16 (16 dwords) x 2 buffers
  __shared__ u32 Bs[2][64 * 16];   // 64 cols  x 32 bf16 x 2 buffers

  const int tid  = threadIdx.x;
  const int wave = tid >> 5;
  const int lane = tid & 31;
  const int lm   = lane & 15;
  const int hi   = lane >> 4;
  const int row0 = blockIdx.y * 128;
  const int col0 = blockIdx.x * 64;
  const int Kd   = K >> 1;         // dwords per bf16 row

  v8f acc[2][4];
#pragma unroll
  for (int mi = 0; mi < 2; mi++)
#pragma unroll
    for (int nt = 0; nt < 4; nt++) acc[mi][nt] = (v8f){};

  const u32* Bg = (const u32*)Bt;

  // stage tiles for K-step k0 into buffer buf
  auto issue = [&](int buf, int k0) {
    // B tile: 4KB = 256 x b128; 2 async per thread
#pragma unroll
    for (int i = 0; i < 2; i++) {
      int t  = tid + i * 128;            // 0..255
      int r  = t >> 2;                   // 0..63
      int c4 = (t & 3) * 4;              // dword offset in row
      u32 ldsoff = (u32)(uintptr_t)&Bs[buf][r * 16 + c4];
      u64 ga = (u64)(const void*)(Bg + (size_t)(col0 + r) * Kd + (k0 >> 1) + c4);
      ASYNC_B128(ldsoff, ga);
    }
    if (A_F32) {
      const float* A = (const float*)Aptr;
#pragma unroll
      for (int it = 0; it < 16; it++) {
        int idx = tid + it * 128;        // 2048 dwords
        int r = idx >> 4, c = idx & 15;
        size_t base = (size_t)(row0 + r) * K + k0 + c * 2;
        float f0 = A[base], f1 = A[base + 1];
        As[buf][r * 16 + c] = (u32)f2bf(f0) | ((u32)f2bf(f1) << 16);
      }
    } else {
      const u32* A = (const u32*)Aptr;
#pragma unroll
      for (int i = 0; i < 4; i++) {
        int t  = tid + i * 128;          // 0..511
        int r  = t >> 2;                 // 0..127
        int c4 = (t & 3) * 4;
        u32 ldsoff = (u32)(uintptr_t)&As[buf][r * 16 + c4];
        u64 ga = (u64)(const void*)(A + (size_t)(row0 + r) * Kd + (k0 >> 1) + c4);
        ASYNC_B128(ldsoff, ga);
      }
    }
  };

  const int steps = K >> 5;
  issue(0, 0);

  for (int s = 0; s < steps; s++) {
    const int cur = s & 1;
    const int nxt = cur ^ 1;
    // WAR: everyone finished computing on 'nxt' (iter s-1) before we overwrite
    __syncthreads();
    if (s + 1 < steps) {
      issue(nxt, (s + 1) * 32);
      // in-order async completion: waiting to <= inflight(next) retires 'cur'
      if (A_F32) asm volatile("s_wait_asynccnt 0x2" ::: "memory");
      else       asm volatile("s_wait_asynccnt 0x6" ::: "memory");
    } else {
      asm volatile("s_wait_asynccnt 0x0" ::: "memory");
    }
    __syncthreads();   // cross-wave visibility of 'cur'

    v16bf a0 = load_frag(As[cur], wave * 32 + lm, hi);
    v16bf a1 = load_frag(As[cur], wave * 32 + 16 + lm, hi);
#pragma unroll
    for (int nt = 0; nt < 4; nt++) {
      v16bf b = load_frag(Bs[cur], nt * 16 + lm, hi);
      acc[0][nt] = __builtin_amdgcn_wmma_f32_16x16x32_bf16(false, a0, false, b, (short)0, acc[0][nt], false, false);
      acc[1][nt] = __builtin_amdgcn_wmma_f32_16x16x32_bf16(false, a1, false, b, (short)0, acc[1][nt], false, false);
    }
  }

  // Epilogue. C/D layout: lane L, vgpr r -> m = base + r + (L>>4)*8, n = nt*16 + (L&15)
#pragma unroll
  for (int mi = 0; mi < 2; mi++) {
#pragma unroll
    for (int nt = 0; nt < 4; nt++) {
#pragma unroll
      for (int r = 0; r < 8; r++) {
        int m = wave * 32 + mi * 16 + r + hi * 8;
        int n = nt * 16 + lm;
        float v = acc[mi][nt][r] + bias[col0 + n];
        size_t off = (size_t)(row0 + m) * Ncols + col0 + n;
        if (EPI == 0) {
          ((float*)Cptr)[off] = v;
        } else {
          v = v > 0.0f ? v : (expf(v) - 1.0f);
          ((u16*)Cptr)[off] = f2bf(v);
        }
      }
    }
  }
}

// -------------------------------------------------------------------------
// Row LayerNorm (D = 256 or 512), optional ELU, writes f32 (optional, may be
// in-place) and bf16 (optional).
// -------------------------------------------------------------------------
__global__ __launch_bounds__(256) void ln_kernel(const float* __restrict__ in,
                                                 const float* __restrict__ g,
                                                 const float* __restrict__ b,
                                                 float* out_f32, u16* out_bf,
                                                 int D, int do_elu) {
  __shared__ float sbuf[256];
  const int tid = threadIdx.x;
  size_t row = blockIdx.x;
  const float* x = in + row * (size_t)D;

  float lsum = 0.0f;
  for (int c = tid; c < D; c += 256) lsum += x[c];
  sbuf[tid] = lsum; __syncthreads();
  for (int s = 128; s > 0; s >>= 1) { if (tid < s) sbuf[tid] += sbuf[tid + s]; __syncthreads(); }
  float mean = sbuf[0] / (float)D;
  __syncthreads();

  float lvar = 0.0f;
  for (int c = tid; c < D; c += 256) { float d = x[c] - mean; lvar += d * d; }
  sbuf[tid] = lvar; __syncthreads();
  for (int s = 128; s > 0; s >>= 1) { if (tid < s) sbuf[tid] += sbuf[tid + s]; __syncthreads(); }
  float inv = rsqrtf(sbuf[0] / (float)D + 1e-5f);

  for (int c = tid; c < D; c += 256) {
    float v = (x[c] - mean) * inv * g[c] + b[c];
    if (out_f32) out_f32[row * (size_t)D + c] = v;
    if (out_bf) {
      float w = v;
      if (do_elu) w = w > 0.0f ? w : (expf(w) - 1.0f);
      out_bf[row * (size_t)D + c] = f2bf(w);
    }
  }
}

// -------------------------------------------------------------------------
// q = query[t-3] @ qkv_w[:, :H] + qkv_b[:H]    (one block, 256 threads)
// -------------------------------------------------------------------------
__global__ __launch_bounds__(256) void q_kernel(const float* __restrict__ query,
                                                const float* __restrict__ qkv_w,
                                                const float* __restrict__ qkv_b,
                                                const int* __restrict__ tptr,
                                                float* __restrict__ q) {
  int j = threadIdx.x;
  int t = *tptr;
  const float* qr = query + (size_t)(t - 3) * HDIM;
  float s = qkv_b[j];
  for (int k = 0; k < HDIM; k++) s += qr[k] * qkv_w[(size_t)k * (3 * HDIM) + j];
  q[j] = s;
}

// -------------------------------------------------------------------------
// Attention: one block per (b, head). kv: [B*N, 512] (k | v).
// ctx: [B, 256] with layout h*32+d.
// -------------------------------------------------------------------------
__global__ __launch_bounds__(256) void attn_kernel(const float* __restrict__ kv,
                                                   const float* __restrict__ q,
                                                   float* __restrict__ ctx) {
  __shared__ float sc[NNODE];
  __shared__ float qh[32];
  __shared__ float red[256];
  const int tid = threadIdx.x;
  const int bh = blockIdx.x;
  const int b = bh >> 3, h = bh & 7;

  if (tid < 32) qh[tid] = q[h * 32 + tid];
  __syncthreads();

  const float scale = 0.17677669529663687f;  // 1/sqrt(32)
  for (int n = tid; n < NNODE; n += 256) {
    const float* kp = kv + ((size_t)b * NNODE + n) * 512 + h * 32;
    float s = 0.0f;
#pragma unroll
    for (int d = 0; d < 32; d++) s += qh[d] * kp[d];
    sc[n] = s * scale;
  }
  __syncthreads();

  float lmax = -1e30f;
  for (int n = tid; n < NNODE; n += 256) lmax = fmaxf(lmax, sc[n]);
  red[tid] = lmax; __syncthreads();
  for (int s = 128; s > 0; s >>= 1) { if (tid < s) red[tid] = fmaxf(red[tid], red[tid + s]); __syncthreads(); }
  float mx = red[0]; __syncthreads();

  float lsum = 0.0f;
  for (int n = tid; n < NNODE; n += 256) { float e = expf(sc[n] - mx); sc[n] = e; lsum += e; }
  red[tid] = lsum; __syncthreads();
  for (int s = 128; s > 0; s >>= 1) { if (tid < s) red[tid] += red[tid + s]; __syncthreads(); }
  float inv = 1.0f / red[0];
  __syncthreads();

  const int d = tid & 31, g = tid >> 5;  // 8 groups over n
  float acc = 0.0f;
  for (int n = g; n < NNODE; n += 8)
    acc += sc[n] * kv[((size_t)b * NNODE + n) * 512 + 256 + h * 32 + d];
  red[d * 8 + g] = acc; __syncthreads();
  if (tid < 32) {
    float s = 0.0f;
#pragma unroll
    for (int gg = 0; gg < 8; gg++) s += red[tid * 8 + gg];
    ctx[((size_t)b * 8 + h) * 32 + tid] = s * inv;
  }
}

// -------------------------------------------------------------------------
// new_feat = ctx @ out_w + out_b    (128 blocks x 256 threads)
// -------------------------------------------------------------------------
__global__ __launch_bounds__(256) void newfeat_kernel(const float* __restrict__ ctx,
                                                      const float* __restrict__ W,
                                                      const float* __restrict__ bias,
                                                      float* __restrict__ out) {
  int b = blockIdx.x, j = threadIdx.x;
  float s = bias[j];
  for (int k = 0; k < HDIM; k++) s += ctx[(size_t)b * HDIM + k] * W[(size_t)k * HDIM + j];
  out[(size_t)b * HDIM + j] = s;
}

// -------------------------------------------------------------------------
// time embedding (dim 512): sin for i<256, cos for i>=256
// -------------------------------------------------------------------------
__global__ void temb_kernel(const int* __restrict__ tptr, float* __restrict__ temb) {
  int i = threadIdx.x;
  float t = (float)(*tptr);
  int j = (i < HDIM) ? i : (i - HDIM);
  float fr = expf(-logf(10000.0f) * (float)j / (float)HDIM);
  temb[i] = (i < HDIM) ? sinf(t * fr) : cosf(t * fr);
}

// -------------------------------------------------------------------------
// e[b,i,:] = concat(new_feat[b], max(x[b,i], x[b,parent])) + temb  -> bf16
// one block per edge row.
// -------------------------------------------------------------------------
__global__ __launch_bounds__(256) void ebuild_kernel(const float* __restrict__ x,
                                                     const int* __restrict__ edge,
                                                     const float* __restrict__ newf,
                                                     const float* __restrict__ temb,
                                                     u16* __restrict__ e) {
  size_t row = blockIdx.x;            // 0 .. B*(N-1)-1
  int b = (int)(row / (NNODE - 1));
  int i = (int)(row - (size_t)b * (NNODE - 1));
  int p = edge[((size_t)b * NNODE + i) * 2 + 0];
  const float* xc  = x + ((size_t)b * NNODE + i) * HDIM;
  const float* xp  = x + ((size_t)b * NNODE + p) * HDIM;
  const float* nfb = newf + (size_t)b * HDIM;
  u16* er = e + row * (2 * HDIM);
  for (int c = threadIdx.x; c < 2 * HDIM; c += 256) {
    float v = (c < HDIM) ? nfb[c] : fmaxf(xc[c - HDIM], xp[c - HDIM]);
    er[c] = f2bf(v + temb[c]);
  }
}

// -------------------------------------------------------------------------
// logits[row] = h2[row,:] . r3_w + r3_b   (h2 bf16, one block per row)
// -------------------------------------------------------------------------
__global__ __launch_bounds__(256) void r3_kernel(const u16* __restrict__ h,
                                                 const float* __restrict__ w,
                                                 const float* __restrict__ b,
                                                 float* __restrict__ logits) {
  __shared__ float red[256];
  const int tid = threadIdx.x;
  size_t row = blockIdx.x;
  float s = 0.0f;
  for (int c = tid; c < 512; c += 256) s += bf2f(h[row * 512 + c]) * w[c];
  red[tid] = s; __syncthreads();
  for (int st = 128; st > 0; st >>= 1) { if (tid < st) red[tid] += red[tid + st]; __syncthreads(); }
  if (tid == 0) logits[row] = red[0] + b[0];
}

// -------------------------------------------------------------------------
// log_softmax over L=1021 per batch
// -------------------------------------------------------------------------
__global__ __launch_bounds__(256) void lsm_kernel(const float* __restrict__ logits,
                                                  float* __restrict__ out, int L) {
  __shared__ float red[256];
  const int tid = threadIdx.x;
  const float* x = logits + (size_t)blockIdx.x * L;
  float* o = out + (size_t)blockIdx.x * L;

  float lmax = -1e30f;
  for (int i = tid; i < L; i += 256) lmax = fmaxf(lmax, x[i]);
  red[tid] = lmax; __syncthreads();
  for (int s = 128; s > 0; s >>= 1) { if (tid < s) red[tid] = fmaxf(red[tid], red[tid + s]); __syncthreads(); }
  float mx = red[0]; __syncthreads();

  float lsum = 0.0f;
  for (int i = tid; i < L; i += 256) lsum += expf(x[i] - mx);
  red[tid] = lsum; __syncthreads();
  for (int s = 128; s > 0; s >>= 1) { if (tid < s) red[tid] += red[tid + s]; __syncthreads(); }
  float lse = mx + logf(red[0]);

  for (int i = tid; i < L; i += 256) o[i] = x[i] - lse;
}

// -------------------------------------------------------------------------
extern "C" void kernel_launch(void* const* d_in, const int* in_sizes, int n_in,
                              void* d_out, int out_size, void* d_ws, size_t ws_size,
                              hipStream_t stream) {
  const float* nf       = (const float*)d_in[0];
  const int*   edge     = (const int*)  d_in[1];
  const int*   tptr     = (const int*)  d_in[2];
  const float* query    = (const float*)d_in[3];
  const float* mlp_w1   = (const float*)d_in[4];
  const float* mlp_b1   = (const float*)d_in[5];
  const float* mlp_w2   = (const float*)d_in[6];
  const float* mlp_b2   = (const float*)d_in[7];
  const float* mlp_ln_g = (const float*)d_in[8];
  const float* mlp_ln_b = (const float*)d_in[9];
  const float* qkv_w    = (const float*)d_in[10];
  const float* qkv_b    = (const float*)d_in[11];
  const float* out_w    = (const float*)d_in[12];
  const float* out_b    = (const float*)d_in[13];
  const float* r1_w     = (const float*)d_in[14];
  const float* r1_b     = (const float*)d_in[15];
  const float* r1_ln_g  = (const float*)d_in[16];
  const float* r1_ln_b  = (const float*)d_in[17];
  const float* r2_w     = (const float*)d_in[18];
  const float* r2_b     = (const float*)d_in[19];
  const float* r2_ln_g  = (const float*)d_in[20];
  const float* r2_ln_b  = (const float*)d_in[21];
  const float* r3_w     = (const float*)d_in[22];
  const float* r3_b     = (const float*)d_in[23];

  const size_t M  = (size_t)BATCH * NNODE;        // 130816 = 128*1022
  const size_t Me = (size_t)BATCH * (NNODE - 1);  // 130688 = 128*1021

  char* ws = (char*)d_ws;
  size_t off = 0;
  auto alloc = [&](size_t bytes) -> size_t {
    size_t o = off;
    off += (bytes + 255) & ~(size_t)255;
    return o;
  };
  size_t o_w1t   = alloc((size_t)512 * 256 * 2);
  size_t o_w2t   = alloc((size_t)256 * 256 * 2);
  size_t o_wkvt  = alloc((size_t)512 * 256 * 2);
  size_t o_r1t   = alloc((size_t)512 * 512 * 2);
  size_t o_r2t   = alloc((size_t)512 * 512 * 2);
  size_t o_q     = alloc((size_t)256 * 4);
  size_t o_ctx   = alloc((size_t)BATCH * HDIM * 4);
  size_t o_nfeat = alloc((size_t)BATCH * HDIM * 4);
  size_t o_temb  = alloc((size_t)512 * 4);
  size_t o_log   = alloc(Me * 4);
  size_t o_R4    = alloc(M * 256 * 2);   // xmid bf16 -> xbf bf16
  size_t o_R3    = alloc(M * 256 * 4);   // xpre/x f32 -> h1 bf16
  size_t o_R1    = alloc(M * 512 * 4);   // kv f32 -> e bf16 -> h2pre f32
  size_t o_R2    = alloc(Me * 512 * 4);  // h1pre f32 -> h2 bf16
  (void)ws_size; (void)n_in; (void)in_sizes; (void)out_size;

  u16* w1t  = (u16*)(ws + o_w1t);
  u16* w2t  = (u16*)(ws + o_w2t);
  u16* wkvt = (u16*)(ws + o_wkvt);
  u16* r1t  = (u16*)(ws + o_r1t);
  u16* r2t  = (u16*)(ws + o_r2t);
  float* qv    = (float*)(ws + o_q);
  float* ctx   = (float*)(ws + o_ctx);
  float* nfeat = (float*)(ws + o_nfeat);
  float* temb  = (float*)(ws + o_temb);
  float* logit = (float*)(ws + o_log);
  u16*   xmid  = (u16*)(ws + o_R4);   // then xbf
  float* xpre  = (float*)(ws + o_R3); // then x in-place, then region reused as h1 bf16
  u16*   h1bf  = (u16*)(ws + o_R3);
  float* kv    = (float*)(ws + o_R1);
  u16*   ebf   = (u16*)(ws + o_R1);
  float* h2pre = (float*)(ws + o_R1);
  float* h1pre = (float*)(ws + o_R2);
  u16*   h2bf  = (u16*)(ws + o_R2);

  // ---- weight conversion (transpose to (N,K) bf16) ----
  convw_kernel<<<(512 * 256 + 255) / 256, 256, 0, stream>>>(mlp_w1, w1t, 512, 256, 256, 0);
  convw_kernel<<<(256 * 256 + 255) / 256, 256, 0, stream>>>(mlp_w2, w2t, 256, 256, 256, 0);
  convw_kernel<<<(256 * 512 + 255) / 256, 256, 0, stream>>>(qkv_w, wkvt, 256, 512, 768, 256);
  convw_kernel<<<(512 * 512 + 255) / 256, 256, 0, stream>>>(r1_w, r1t, 512, 512, 512, 0);
  convw_kernel<<<(512 * 512 + 255) / 256, 256, 0, stream>>>(r2_w, r2t, 512, 512, 512, 0);
  temb_kernel<<<1, 512, 0, stream>>>(tptr, temb);
  q_kernel<<<1, 256, 0, stream>>>(query, qkv_w, qkv_b, tptr, qv);

  // ---- GEMM1: xmid = elu(nf @ w1 + b1)  [bf16] ----
  gemm_wmma<true, 1><<<dim3(256 / 64, (unsigned)(M / 128)), 128, 0, stream>>>(
      (const void*)nf, w1t, mlp_b1, (void*)xmid, 512, 256);
  // ---- GEMM2: xpre = xmid @ w2 + b2  [f32] ----
  gemm_wmma<false, 0><<<dim3(256 / 64, (unsigned)(M / 128)), 128, 0, stream>>>(
      (const void*)xmid, w2t, mlp_b2, (void*)xpre, 256, 256);
  // ---- LN: x (f32, in place) + xbf (bf16, reuses xmid region) ----
  ln_kernel<<<(unsigned)M, 256, 0, stream>>>(xpre, mlp_ln_g, mlp_ln_b, xpre, xmid, 256, 0);
  // ---- GEMM3: kv = xbf @ Wkv + qkv_b[H:]  [f32] ----
  gemm_wmma<false, 0><<<dim3(512 / 64, (unsigned)(M / 128)), 128, 0, stream>>>(
      (const void*)xmid, wkvt, qkv_b + 256, (void*)kv, 256, 512);
  // ---- attention + projection ----
  attn_kernel<<<BATCH * 8, 256, 0, stream>>>(kv, qv, ctx);
  newfeat_kernel<<<BATCH, 256, 0, stream>>>(ctx, out_w, out_b, nfeat);
  // ---- edge build: e bf16 (reuses kv region) ----
  ebuild_kernel<<<(unsigned)Me, 256, 0, stream>>>(xpre, edge, nfeat, temb, ebf);
  // ---- r1: h1pre = e @ r1 + b  [f32] ----
  gemm_wmma<false, 0><<<dim3(512 / 64, (unsigned)(Me / 128)), 128, 0, stream>>>(
      (const void*)ebf, r1t, r1_b, (void*)h1pre, 512, 512);
  ln_kernel<<<(unsigned)Me, 256, 0, stream>>>(h1pre, r1_ln_g, r1_ln_b, (float*)nullptr, h1bf, 512, 1);
  // ---- r2: h2pre = h1 @ r2 + b  [f32] (reuses R1) ----
  gemm_wmma<false, 0><<<dim3(512 / 64, (unsigned)(Me / 128)), 128, 0, stream>>>(
      (const void*)h1bf, r2t, r2_b, (void*)h2pre, 512, 512);
  ln_kernel<<<(unsigned)Me, 256, 0, stream>>>(h2pre, r2_ln_g, r2_ln_b, (float*)nullptr, h2bf, 512, 1);
  // ---- r3 + log_softmax ----
  r3_kernel<<<(unsigned)Me, 256, 0, stream>>>(h2bf, r3_w, r3_b, logit);
  lsm_kernel<<<BATCH, 256, 0, stream>>>(logit, (float*)d_out, NNODE - 1);
}
